// MHA_55705725829792
// MI455X (gfx1250) — matrix operand order repaired and checked by reading, not compiled, not run on hardware
//
#include <hip/hip_runtime.h>
#include <hip/hip_bf16.h>

// ---------------------------------------------------------------------------
// MHA graph-attention pipeline for MI455X (gfx1250), wave32 + WMMA f16 + TDM.
//   B=16, Tq=512, Tkv=513, C=1024, H=16, D=64
// Pipeline:
//   1) transpose_cast : W (f32 [K][N]) -> W^T (f16 [N][K])   (LDS tiled)
//   2) ln_cast        : LayerNorm(x) -> f16
//   3) adj_to_bits    : adjacency int32 -> bitmask words (col 512 forced on)
//   4) gemm_f16w      : f16 x f16 -> f16/f32 via v_wmma_f32_16x16x32_f16,
//                       A/B tiles staged to LDS by the Tensor Data Mover
//                       (tensor_load_to_lds, double-buffered, s_wait_tensorcnt),
//                       HW row padding -> bank-conflict-free ds_load_b128.
//   5) v_transpose    : V -> V^T [B,H,D,Tkv_pad] (zero-padded)
//   6) attn           : flash attention, S^T trick (P fragment = in-lane)
//   7) gemm_f16w<float>: Y @ Wc^T -> d_out (fp32)
// ---------------------------------------------------------------------------

#define HID   1024
#define NH    16
#define DH    64
#define TQ    512
#define TKV   513
#define TKV_PAD 544
#define NB    16
#define ADJW  17   // 544/32 mask words per q row

#define LDS_STRIDE 40   // 32 data halves + 8 pad halves (16B) per tile row

typedef __attribute__((ext_vector_type(16))) _Float16 v16h;
typedef __attribute__((ext_vector_type(8)))  float    v8f;
typedef __attribute__((ext_vector_type(4)))  unsigned v4u;
typedef __attribute__((ext_vector_type(8)))  int      v8i;
typedef __attribute__((ext_vector_type(4)))  int      v4i;

union FragU {
    v16h  h;
    float4 f4[2];
};

#if defined(__HIP_DEVICE_COMPILE__) && __has_builtin(__builtin_amdgcn_tensor_load_to_lds)
#define USE_TDM 1
#else
#define USE_TDM 0
#endif

__device__ __forceinline__ v8f wmma_f16(v16h a, v16h b, v8f c) {
    // D = A(16x32) * B(32x16) + C, fp32 accumulate
    return __builtin_amdgcn_wmma_f32_16x16x32_f16(
        /*neg_a=*/false, a, /*neg_b=*/false, b,
        /*c_mod=*/(short)0, c, /*reuse_a=*/false, /*reuse_b=*/false);
}

#if USE_TDM
// Issue one TDM 2D tile load: 32 halves x 128 rows from a row-major f16
// tensor (row stride = Kld elements) into LDS at byte offset lds_off.
// HW padding: 16 DWORDs (one 64B row) then 4 DWORDs pad -> 80B LDS row stride.
// rows_valid < 128 rows  => TDM returns zeros for OOB rows (M tail).
__device__ __forceinline__ void tdm_load_2d(const _Float16* gbase,
                                            unsigned lds_off,
                                            int rows_valid, int Kld) {
    const unsigned long long ga = (unsigned long long)(size_t)gbase;
    v4u g0;
    g0[0] = 1u;                                        // count=1 (valid), user mode
    g0[1] = lds_off;                                   // LDS byte address
    g0[2] = (unsigned)(ga & 0xFFFFFFFFu);              // global_addr[31:0]
    g0[3] = (unsigned)((ga >> 32) & 0x1FFFFFFu)        // global_addr[56:32]
          | (2u << 30);                                // type = 2 ("image")
    v8i g1;
    g1[0] = (1 << 16)                                  // data_size = 2 bytes
          | (1 << 20)                                  // pad_enable
          | (3 << 22)                                  // pad_interval: 16 DWORDs
          | (3 << 25);                                 // pad_amount: 4 DWORDs
    g1[1] = (int)(32u << 16);                          // tensor_dim0[15:0] = 32
    g1[2] = (int)(((unsigned)rows_valid & 0xFFFFu) << 16); // dim0 hi=0 | dim1 lo
    g1[3] = (int)((((unsigned)rows_valid >> 16) & 0xFFFFu) // dim1 hi
          | (32u << 16));                              // tile_dim0 = 32
    g1[4] = 128;                                       // tile_dim1=128, tile_dim2=0
    g1[5] = Kld;                                       // dim0_stride[31:0] = K elems
    g1[6] = 0;                                         // dim0_stride hi, dim1_stride lo
    g1[7] = 0;
    v4i z4 = {};
    v8i z8 = {};
    __builtin_amdgcn_tensor_load_to_lds(g0, g1, z4, z4, z8, 0);
}

__device__ __forceinline__ unsigned lds_byte_off(const void* p) {
    return (unsigned)(size_t)p;   // generic LDS VA: low 32 bits = LDS offset
}
#endif

// ---------------------------------------------------------------------------
// 1) Weight transpose + cast: W f32 [K][N] -> WT f16 [N][K]
// ---------------------------------------------------------------------------
__global__ void __launch_bounds__(256)
transpose_cast(const float* __restrict__ W, _Float16* __restrict__ WT,
               int K, int N) {
    __shared__ float t[16][17];
    const int tx = threadIdx.x & 15;
    const int ty = threadIdx.x >> 4;
    const int n0 = blockIdx.x * 16;
    const int k0 = blockIdx.y * 16;
    t[ty][tx] = W[(size_t)(k0 + ty) * N + n0 + tx];
    __syncthreads();
    WT[(size_t)(n0 + ty) * K + k0 + tx] = (_Float16)t[tx][ty];
}

// ---------------------------------------------------------------------------
// 2) LayerNorm + cast to f16. One block (256 thr) per row of HID=1024.
// ---------------------------------------------------------------------------
__global__ void __launch_bounds__(256)
ln_cast(const float* __restrict__ x, const float* __restrict__ g,
        const float* __restrict__ bta, _Float16* __restrict__ out) {
    const int row = blockIdx.x;
    const int tid = threadIdx.x;
    const float4 v = reinterpret_cast<const float4*>(x + (size_t)row * HID)[tid];
    float s = v.x + v.y + v.z + v.w;
    float q = v.x * v.x + v.y * v.y + v.z * v.z + v.w * v.w;
#pragma unroll
    for (int off = 16; off > 0; off >>= 1) {
        s += __shfl_xor(s, off, 32);
        q += __shfl_xor(q, off, 32);
    }
    __shared__ float ss[8], qq[8];
    const int wave = tid >> 5, lane = tid & 31;
    if (lane == 0) { ss[wave] = s; qq[wave] = q; }
    __syncthreads();
    s = 0.f; q = 0.f;
#pragma unroll
    for (int i = 0; i < 8; ++i) { s += ss[i]; q += qq[i]; }
    const float mu  = s * (1.0f / HID);
    const float var = q * (1.0f / HID) - mu * mu;
    const float rs  = rsqrtf(var + 1e-5f);
    const float4 gv = reinterpret_cast<const float4*>(g)[tid];
    const float4 bv = reinterpret_cast<const float4*>(bta)[tid];
    _Float16* o = out + (size_t)row * HID + tid * 4;
    o[0] = (_Float16)((v.x - mu) * rs * gv.x + bv.x);
    o[1] = (_Float16)((v.y - mu) * rs * gv.y + bv.y);
    o[2] = (_Float16)((v.z - mu) * rs * gv.z + bv.z);
    o[3] = (_Float16)((v.w - mu) * rs * gv.w + bv.w);
}

// ---------------------------------------------------------------------------
// 3) adjacency -> bitmask. word t of row q covers kv [32t, 32t+31].
//    word 16: only bit 0 set (the always-attendable extra token, kv=512).
// ---------------------------------------------------------------------------
__global__ void __launch_bounds__(32)
adj_to_bits(const int* __restrict__ adj, unsigned* __restrict__ bits) {
    const int q = blockIdx.x;
    const int t = threadIdx.x;
    if (t >= ADJW) return;
    unsigned w = 0u;
    if (t < 16) {
        const int* row = adj + (size_t)q * 512 + t * 32;
#pragma unroll
        for (int j = 0; j < 32; ++j) w |= (row[j] ? 1u : 0u) << j;
    } else {
        w = 1u;
    }
    bits[(size_t)q * ADJW + t] = w;
}

// ---------------------------------------------------------------------------
// 4) WMMA GEMM: C[M][N] = A[M][K] * B[K][N], B given transposed (BT [N][K]).
//    Block = 256 thr = 8 waves (4 along M x 2 along N). Block tile 128x128,
//    wave tile 32x64, K step 32 -> 8 v_wmma per wave per step.
//    A/B 128x32 tiles staged in LDS by TDM (double-buffered).
// ---------------------------------------------------------------------------
template <typename OutT>
__global__ void __launch_bounds__(256)
gemm_f16w(const _Float16* __restrict__ A, const _Float16* __restrict__ BT,
          OutT* __restrict__ C, int M, int N, int K) {
    __shared__ _Float16 sA[2][128 * LDS_STRIDE];
    __shared__ _Float16 sB[2][128 * LDS_STRIDE];
    const int lane = threadIdx.x & 31;
    const int wave = threadIdx.x >> 5;
    const int ln = lane & 15;
    const int hf = lane >> 4;
    const int m0 = blockIdx.y * 128;
    const int n0 = blockIdx.x * 128;
    const int wm = (wave & 3) * 32;   // wave row offset in block tile
    const int wn = (wave >> 2) * 64;  // wave col offset in block tile

    v8f acc[2][4] = {};

#if USE_TDM
    const int mrem = M - m0;          // TDM zero-fills rows >= mrem
    if (wave == 0) {
        tdm_load_2d(A + (size_t)m0 * K, lds_byte_off(&sA[0][0]), mrem, K);
        tdm_load_2d(BT + (size_t)n0 * K, lds_byte_off(&sB[0][0]), 128, K);
    }
    int buf = 0;
    for (int kk = 0; kk < K; kk += 32) {
        if (wave == 0) {
            if (kk + 32 < K) {
                tdm_load_2d(A + (size_t)m0 * K + kk + 32,
                            lds_byte_off(&sA[buf ^ 1][0]), mrem, K);
                tdm_load_2d(BT + (size_t)n0 * K + kk + 32,
                            lds_byte_off(&sB[buf ^ 1][0]), 128, K);
                __builtin_amdgcn_s_wait_tensorcnt(2);  // current bufs complete
            } else {
                __builtin_amdgcn_s_wait_tensorcnt(0);
            }
        }
        __syncthreads();
        FragU a[2], b[4];
#pragma unroll
        for (int mi = 0; mi < 2; ++mi) {
            const _Float16* p = &sA[buf][(wm + mi * 16 + ln) * LDS_STRIDE + hf * 8];
            a[mi].f4[0] = *reinterpret_cast<const float4*>(p);       // K {0..7}/{8..15}
            a[mi].f4[1] = *reinterpret_cast<const float4*>(p + 16);  // K {16..23}/{24..31}
        }
#pragma unroll
        for (int ni = 0; ni < 4; ++ni) {
            const _Float16* p = &sB[buf][(wn + ni * 16 + ln) * LDS_STRIDE + hf * 8];
            b[ni].f4[0] = *reinterpret_cast<const float4*>(p);       // K {0..7}/{8..15}
            b[ni].f4[1] = *reinterpret_cast<const float4*>(p + 16);  // K {16..23}/{24..31}
        }
#pragma unroll
        for (int mi = 0; mi < 2; ++mi)
#pragma unroll
            for (int ni = 0; ni < 4; ++ni)
                acc[mi][ni] = wmma_f16(a[mi].h, b[ni].h, acc[mi][ni]);
        __syncthreads();
        buf ^= 1;
    }
#else
    // Fallback staging: plain global loads + LDS stores, single buffer.
    for (int kk = 0; kk < K; kk += 32) {
#pragma unroll
        for (int i = 0; i < 2; ++i) {
            const int t = threadIdx.x + i * 256;
            const int row = t >> 2;
            const int c = (t & 3) * 8;
            int gr = m0 + row; if (gr >= M) gr = M - 1;
            const float4 va = *reinterpret_cast<const float4*>(
                A + (size_t)gr * K + kk + c);
            *reinterpret_cast<float4*>(&sA[0][row * LDS_STRIDE + c]) = va;
            const float4 vb = *reinterpret_cast<const float4*>(
                BT + (size_t)(n0 + row) * K + kk + c);
            *reinterpret_cast<float4*>(&sB[0][row * LDS_STRIDE + c]) = vb;
        }
        __syncthreads();
        FragU a[2], b[4];
#pragma unroll
        for (int mi = 0; mi < 2; ++mi) {
            const _Float16* p = &sA[0][(wm + mi * 16 + ln) * LDS_STRIDE + hf * 8];
            a[mi].f4[0] = *reinterpret_cast<const float4*>(p);
            a[mi].f4[1] = *reinterpret_cast<const float4*>(p + 16);
        }
#pragma unroll
        for (int ni = 0; ni < 4; ++ni) {
            const _Float16* p = &sB[0][(wn + ni * 16 + ln) * LDS_STRIDE + hf * 8];
            b[ni].f4[0] = *reinterpret_cast<const float4*>(p);
            b[ni].f4[1] = *reinterpret_cast<const float4*>(p + 16);
        }
#pragma unroll
        for (int mi = 0; mi < 2; ++mi)
#pragma unroll
            for (int ni = 0; ni < 4; ++ni)
                acc[mi][ni] = wmma_f16(a[mi].h, b[ni].h, acc[mi][ni]);
        __syncthreads();
    }
#endif

#pragma unroll
    for (int mi = 0; mi < 2; ++mi) {
#pragma unroll
        for (int ni = 0; ni < 4; ++ni) {
            const int rbase = m0 + wm + mi * 16 + hf * 8;
            const int col = n0 + wn + ni * 16 + ln;
#pragma unroll
            for (int r = 0; r < 8; ++r) {
                const int row = rbase + r;
                if (row < M) C[(size_t)row * N + col] = (OutT)acc[mi][ni][r];
            }
        }
    }
}

// ---------------------------------------------------------------------------
// 5) V transpose: kvp[b*TKV+kv][1024 + h*64 + d] -> VT[(b*NH+h)*64+d][kv],
//    zero-padded to TKV_PAD so attention B-fragments are plain b128 loads.
// ---------------------------------------------------------------------------
__global__ void __launch_bounds__(256)
v_transpose(const _Float16* __restrict__ kvp, _Float16* __restrict__ VT) {
    __shared__ _Float16 t[16][17];
    const int tx = threadIdx.x & 15;
    const int ty = threadIdx.x >> 4;
    const int kv0 = blockIdx.x * 16;
    const int d0  = blockIdx.y * 16;
    const int bh  = blockIdx.z;
    const int b = bh >> 4, h = bh & 15;
    const int kv = kv0 + ty;
    _Float16 v = (_Float16)0.0f;
    if (kv < TKV)
        v = kvp[(size_t)(b * TKV + kv) * 2048 + 1024 + h * DH + d0 + tx];
    t[ty][tx] = v;
    __syncthreads();
    VT[((size_t)bh * DH + d0 + ty) * TKV_PAD + kv0 + tx] = t[tx][ty];
}

__global__ void zero_tail(_Float16* __restrict__ p, int n) {
    const int i = blockIdx.x * blockDim.x + threadIdx.x;
    if (i < n) p[i] = (_Float16)0.0f;
}

// ---------------------------------------------------------------------------
// 6) Flash attention. One wave per (16-q tile, b, h).
//    S^T = K * Q^T  ->  S^T accumulator layout == P A-fragment layout,
//    so softmax -> P needs only in-lane f32->f16 converts.
// ---------------------------------------------------------------------------
__global__ void __launch_bounds__(32)
attn(const _Float16* __restrict__ Qp,   // [B*TQ][HID] f16
     const _Float16* __restrict__ Kp,   // kvp [B*TKV(+pad)][2048], k = cols 0..1023
     const _Float16* __restrict__ VT,   // [B*NH][DH][TKV_PAD] f16
     const unsigned* __restrict__ bits, // [TQ][ADJW]
     _Float16* __restrict__ Y) {        // [B*TQ][HID] f16
    const int lane = threadIdx.x;
    const int ln = lane & 15;
    const int hf = lane >> 4;
    const int bh = blockIdx.y;
    const int b = bh >> 4, h = bh & 15;
    const int q0 = blockIdx.x * 16;

    // Q^T B-fragments (reused for all kv steps)
    FragU qt[2];
    const _Float16* qrow = Qp + (size_t)(b * TQ + q0 + ln) * HID + h * DH;
#pragma unroll
    for (int dc = 0; dc < 2; ++dc) {
        const float4* p = reinterpret_cast<const float4*>(qrow + dc * 32 + hf * 8);
        qt[dc].f4[0] = p[0];
        qt[dc].f4[1] = p[2];
    }

    v8f o[4] = {};
    float m_i = -1e30f, l_i = 0.0f;
    const unsigned* brow = bits + (size_t)(q0 + ln) * ADJW;

    for (int t = 0; t < ADJW; ++t) {
        const int kv0 = t * 32;
        // K A-fragments: rows kv0..kv0+31, d split per WMMA A layout
        FragU ka[2][2];
#pragma unroll
        for (int sub = 0; sub < 2; ++sub) {
            const _Float16* kr =
                Kp + (size_t)(b * TKV + kv0 + sub * 16 + ln) * 2048 + h * DH + hf * 8;
#pragma unroll
            for (int dc = 0; dc < 2; ++dc) {
                const float4* p = reinterpret_cast<const float4*>(kr + dc * 32);
                ka[sub][dc].f4[0] = p[0];
                ka[sub][dc].f4[1] = p[2];
            }
        }
        // S^T tiles (M=kv, N=q), K over d=64 -> 2 wmma each
        v8f s0 = {}, s1 = {};
        s0 = wmma_f16(ka[0][0].h, qt[0].h, s0);
        s0 = wmma_f16(ka[0][1].h, qt[1].h, s0);
        s1 = wmma_f16(ka[1][0].h, qt[0].h, s1);
        s1 = wmma_f16(ka[1][1].h, qt[1].h, s1);

        // mask + scale (1/sqrt(64)); sentinel -1e30 keeps softmax NaN-free
        const unsigned w = brow[t];
        float sv0[8], sv1[8];
        float mx = -1e30f;
#pragma unroll
        for (int r = 0; r < 8; ++r) {
            const int b0 = r + hf * 8;
            sv0[r] = ((w >> b0) & 1u)        ? s0[r] * 0.125f : -1e30f;
            sv1[r] = ((w >> (16 + b0)) & 1u) ? s1[r] * 0.125f : -1e30f;
            mx = fmaxf(mx, fmaxf(sv0[r], sv1[r]));
        }
        mx = fmaxf(mx, __shfl_xor(mx, 16, 32));
        const float m_new = fmaxf(m_i, mx);
        const float alpha = __expf(m_i - m_new);

        // P fragment: S^T accumulator maps 1:1 onto the PV A-fragment
        FragU pa;
        float lsum = 0.0f;
#pragma unroll
        for (int r = 0; r < 8; ++r) {
            const float p0 = __expf(sv0[r] - m_new);
            const float p1 = __expf(sv1[r] - m_new);
            lsum += p0 + p1;
            pa.h[r]     = (_Float16)p0;
            pa.h[r + 8] = (_Float16)p1;
        }
        lsum += __shfl_xor(lsum, 16, 32);
        l_i = l_i * alpha + lsum;
        m_i = m_new;

        // rescale O accumulators (per-q factor broadcast by lane)
#pragma unroll
        for (int r = 0; r < 8; ++r) {
            const float alo = __shfl(alpha, r, 32);
            const float ahi = __shfl(alpha, r + 8, 32);
            const float sc = hf ? ahi : alo;
            o[0][r] *= sc; o[1][r] *= sc; o[2][r] *= sc; o[3][r] *= sc;
        }

        // PV: B-fragments from V^T (kv contiguous per lane)
#pragma unroll
        for (int dn = 0; dn < 4; ++dn) {
            FragU vb;
            const float4* p = reinterpret_cast<const float4*>(
                VT + ((size_t)bh * DH + dn * 16 + ln) * TKV_PAD + kv0 + hf * 8);
            vb.f4[0] = p[0];
            vb.f4[1] = p[2];
            o[dn] = wmma_f16(pa.h, vb.h, o[dn]);
        }
    }

    // normalize + store Y (f16)
#pragma unroll
    for (int r = 0; r < 8; ++r) {
        const float ilo = __shfl(l_i, r, 32);
        const float ihi = __shfl(l_i, r + 8, 32);
        const float inv = 1.0f / (hf ? ihi : ilo);
        const int qr = q0 + r + hf * 8;
        _Float16* yo = Y + (size_t)(b * TQ + qr) * HID + h * DH;
#pragma unroll
        for (int dn = 0; dn < 4; ++dn)
            yo[dn * 16 + ln] = (_Float16)(o[dn][r] * inv);
    }
}

// ---------------------------------------------------------------------------
// Launcher
// ---------------------------------------------------------------------------
extern "C" void kernel_launch(void* const* d_in, const int* in_sizes, int n_in,
                              void* d_out, int out_size, void* d_ws, size_t ws_size,
                              hipStream_t stream) {
    (void)in_sizes; (void)n_in; (void)out_size; (void)ws_size;
    const float* q   = (const float*)d_in[0];
    const float* kv  = (const float*)d_in[1];
    const int*   adj = (const int*)d_in[2];
    const float* Wq  = (const float*)d_in[3];
    const float* Wkv = (const float*)d_in[4];
    const float* Wc  = (const float*)d_in[5];
    const float* lng = (const float*)d_in[6];
    const float* lnb = (const float*)d_in[7];
    float* out = (float*)d_out;

    char* ws = (char*)d_ws;
    auto carve = [&](size_t bytes) -> char* {
        char* p = ws;
        ws += (bytes + 255) & ~(size_t)255;
        return p;
    };
    const size_t MQ = (size_t)NB * TQ;    // 8192
    const size_t MK = (size_t)NB * TKV;   // 8208
    _Float16* WqT  = (_Float16*)carve((size_t)1024 * 1024 * 2);
    _Float16* WkvT = (_Float16*)carve((size_t)2048 * 1024 * 2);
    _Float16* WcT  = (_Float16*)carve((size_t)1024 * 1024 * 2);
    _Float16* qn   = (_Float16*)carve(MQ * HID * 2);           // later reused as Y
    _Float16* kvn  = (_Float16*)carve(MK * HID * 2);
    _Float16* Qp   = (_Float16*)carve(MQ * HID * 2);
    _Float16* KVp  = (_Float16*)carve((MK + 32) * 2048 * 2);   // +pad rows
    _Float16* VT   = (_Float16*)carve((size_t)NB * NH * DH * TKV_PAD * 2);
    unsigned* bitsW = (unsigned*)carve((size_t)TQ * ADJW * 4);

    // weights: transpose+cast
    transpose_cast<<<dim3(1024 / 16, 1024 / 16), 256, 0, stream>>>(Wq,  WqT,  1024, 1024);
    transpose_cast<<<dim3(2048 / 16, 1024 / 16), 256, 0, stream>>>(Wkv, WkvT, 1024, 2048);
    transpose_cast<<<dim3(1024 / 16, 1024 / 16), 256, 0, stream>>>(Wc,  WcT,  1024, 1024);
    // layernorm + cast
    ln_cast<<<(int)MQ, 256, 0, stream>>>(q,  lng, lnb, qn);
    ln_cast<<<(int)MK, 256, 0, stream>>>(kv, lng, lnb, kvn);
    // adjacency bitmask
    adj_to_bits<<<TQ, 32, 0, stream>>>(adj, bitsW);
    // projections (WMMA + TDM staging)
    gemm_f16w<_Float16><<<dim3(1024 / 128, (int)(MQ / 128)), 256, 0, stream>>>(
        qn, WqT, Qp, (int)MQ, 1024, 1024);
    gemm_f16w<_Float16><<<dim3(2048 / 128, (int)((MK + 127) / 128)), 256, 0, stream>>>(
        kvn, WkvT, KVp, (int)MK, 2048, 1024);
    // zero pad rows after last batch (read by attention kv tail, masked out)
    zero_tail<<<(32 * 2048 + 255) / 256, 256, 0, stream>>>(KVp + MK * 2048, 32 * 2048);
    // V^T for the PV matmul
    v_transpose<<<dim3(TKV_PAD / 16, DH / 16, NB * NH), 256, 0, stream>>>(KVp, VT);
    // flash attention (WMMA), Y reuses qn storage
    _Float16* Y = qn;
    attn<<<dim3(TQ / 16, NB * NH), 32, 0, stream>>>(Qp, KVp, VT, bitsW, Y);
    // output projection to fp32
    gemm_f16w<float><<<dim3(1024 / 128, (int)(MQ / 128)), 256, 0, stream>>>(
        Y, WcT, out, (int)MQ, 1024, 1024);
}
